// KnnConvUnit_42872363548712
// MI455X (gfx1250) — compile-verified, matrix operand order checked by reading, not run on hardware
//
#include <hip/hip_runtime.h>

// ---------------------------------------------------------------------------
// KnnConvUnit for MI455X (gfx1250): per-point [16,192]x[192,128] +
// [16,128]x[128,128] with relu, k-maxpool, then batched [16,128]x[128,128],
// all on V_WMMA_F32_16X16X32_F16 (f16 in, f32 accumulate).
// Fragment operands are fetched from LDS with ds_load_b128 (2 per fragment).
// ---------------------------------------------------------------------------

typedef __attribute__((ext_vector_type(16))) _Float16 v16h;
typedef __attribute__((ext_vector_type(8)))  float    v8f;
typedef __attribute__((ext_vector_type(4)))  unsigned v4u;

#define NPTS  8192
#define CCH   64
#define KNN   16
#define HIDN  128
#define IN3C  192

// LDS layout (bytes)
#define W1T_OFF   0                       // [128][192] f16 = 49152
#define W2T_OFF   49152                   // [128][128] f16 = 32768
#define W3T_OFF   81920                   // [128][128] f16 = 32768
#define B1_OFF    114688                  // [128] f32
#define B2_OFF    115200
#define B3_OFF    115712
#define WAVE_OFF  116224                  // per-wave: h1 [16][128] f16 + pooled [16][128] f16
#define WAVE_STRIDE 8192
#define SMEM_BYTES (WAVE_OFF + 4 * WAVE_STRIDE)   // 148992 B (< 320KB WGP LDS)

struct FragU { union { v16h v; _Float16 e[16]; unsigned u[8]; v4u q[2]; }; };

// Load one 16x32 f16 WMMA operand striped per the CDNA5 layout from a
// row-major [rows][kdim] f16 array in LDS. "rc" = this lane's row/column,
// "ko" = K offset (multiple of 32), "half" = lane>=16.
// Lane half 0 covers K = ko + {0..7, 16..23}; half 1 covers K + 8.
// The two 8-half groups are contiguous and 16B aligned -> 2x ds_load_b128.
__device__ __forceinline__ v16h loadMat16x32(const _Float16* base_h, int kdim,
                                             int rc, int ko, int half) {
  const v4u* p = (const v4u*)(base_h + rc * kdim + ko + half * 8);
  FragU f;
  f.q[0] = p[0];   // K = ko + half*8 + {0..7}
  f.q[1] = p[2];   // K = ko + half*8 + 16 + {0..7}
  return f.v;
}

__device__ __forceinline__ void packFrag(FragU& fr, const float4 a, const float4 b,
                                         const float4 c, const float4 d) {
  fr.e[0]  = (_Float16)a.x; fr.e[1]  = (_Float16)a.y; fr.e[2]  = (_Float16)a.z; fr.e[3]  = (_Float16)a.w;
  fr.e[4]  = (_Float16)b.x; fr.e[5]  = (_Float16)b.y; fr.e[6]  = (_Float16)b.z; fr.e[7]  = (_Float16)b.w;
  fr.e[8]  = (_Float16)c.x; fr.e[9]  = (_Float16)c.y; fr.e[10] = (_Float16)c.z; fr.e[11] = (_Float16)c.w;
  fr.e[12] = (_Float16)d.x; fr.e[13] = (_Float16)d.y; fr.e[14] = (_Float16)d.z; fr.e[15] = (_Float16)d.w;
}

__device__ __forceinline__ float4 sub4(const float4 a, const float4 b) {
  return make_float4(a.x - b.x, a.y - b.y, a.z - b.z, a.w - b.w);
}

__global__ void __launch_bounds__(128, 1)
knnconv_wmma_kernel(const float* __restrict__ f, const int* __restrict__ knn,
                    const float* __restrict__ W1, const float* __restrict__ b1,
                    const float* __restrict__ W2, const float* __restrict__ b2,
                    const float* __restrict__ W3, const float* __restrict__ b3,
                    float* __restrict__ out) {
  extern __shared__ char smem[];
  _Float16* W1t = (_Float16*)(smem + W1T_OFF);
  _Float16* W2t = (_Float16*)(smem + W2T_OFF);
  _Float16* W3t = (_Float16*)(smem + W3T_OFF);
  float* b1s = (float*)(smem + B1_OFF);
  float* b2s = (float*)(smem + B2_OFF);
  float* b3s = (float*)(smem + B3_OFF);

  const int tid = threadIdx.x;

  // Stage f16 *transposed* weights ([N][K] row-major) so B fragments are
  // contiguous 16B-aligned groups in LDS. Weights are tiny and L2-resident
  // across blocks, so the per-block conversion reads hit L2.
  for (int i = tid; i < IN3C * HIDN; i += 128) {
    int k = i >> 7, n = i & 127;
    W1t[n * IN3C + k] = (_Float16)W1[i];
  }
  for (int i = tid; i < HIDN * HIDN; i += 128) {
    int k = i >> 7, n = i & 127;
    W2t[n * HIDN + k] = (_Float16)W2[i];
    W3t[n * HIDN + k] = (_Float16)W3[i];
  }
  b1s[tid] = b1[tid];
  b2s[tid] = b2[tid];
  b3s[tid] = b3[tid];
  __syncthreads();

  const int wv   = tid >> 5;
  const int lane = tid & 31;
  const int half = lane >> 4;   // which 16-lane half
  const int c16  = lane & 15;   // row (A) / column (B,C,D) index

  _Float16* h1w   = (_Float16*)(smem + WAVE_OFF + wv * WAVE_STRIDE); // [16][128]
  _Float16* poolw = h1w + 16 * HIDN;                                 // [16][128]

  const int waveG = blockIdx.x * 4 + wv;     // 2048 waves
  const int base  = waveG * 16;              // 16 points per wave
  const int b     = base >> 13;              // N = 8192
  const int n0    = base & (NPTS - 1);

  for (int p = 0; p < 16; ++p) {
    const int n = n0 + p;
    // ---- gather: each lane owns neighbor row (c16); lane half picks channel
    //      subset matching the WMMA A layout, so no LDS staging of x.
    const float4* cp = (const float4*)(f + ((size_t)b * NPTS + n) * CCH);
    const int nbidx  = knn[((size_t)b * NPTS + n) * KNN + c16];
    const float4* np = (const float4*)(f + ((size_t)b * NPTS + nbidx) * CCH);

    float4 C[8], Nb[8];
#pragma unroll
    for (int s = 0; s < 4; ++s) {
      C[2 * s]      = cp[s * 4 + half * 2];
      C[2 * s + 1]  = cp[s * 4 + half * 2 + 1];
      Nb[2 * s]     = np[s * 4 + half * 2];
      Nb[2 * s + 1] = np[s * 4 + half * 2 + 1];
    }
    FragU a[6];
    packFrag(a[0], C[0], C[1], C[2], C[3]);       // x cols   0.. 63 (center)
    packFrag(a[1], C[4], C[5], C[6], C[7]);
    packFrag(a[2], Nb[0], Nb[1], Nb[2], Nb[3]);   // x cols  64..127 (neighbor)
    packFrag(a[3], Nb[4], Nb[5], Nb[6], Nb[7]);
    float4 D[8];
#pragma unroll
    for (int s = 0; s < 8; ++s) D[s] = sub4(Nb[s], C[s]);
    packFrag(a[4], D[0], D[1], D[2], D[3]);       // x cols 128..191 (diff)
    packFrag(a[5], D[4], D[5], D[6], D[7]);

    // ---- layer 1: [16,192] x [192,128], relu, -> h1 in LDS
#pragma unroll
    for (int t = 0; t < 8; ++t) {
      v8f acc = {};
#pragma unroll
      for (int kk = 0; kk < 6; ++kk) {
        v16h bf = loadMat16x32(W1t, IN3C, t * 16 + c16, 32 * kk, half);
        acc = __builtin_amdgcn_wmma_f32_16x16x32_f16(false, a[kk].v, false, bf,
                                                     (short)0, acc, false, false);
      }
      const float bias = b1s[t * 16 + c16];
#pragma unroll
      for (int r = 0; r < 8; ++r) {
        float hv = acc[r] + bias;
        hv = hv > 0.f ? hv : 0.f;
        h1w[(half * 8 + r) * HIDN + t * 16 + c16] = (_Float16)hv;
      }
    }

    // ---- layer 2: [16,128] x [128,128], relu, maxpool over the 16 rows
    FragU a2[4];
#pragma unroll
    for (int kk = 0; kk < 4; ++kk)
      a2[kk].v = loadMat16x32(h1w, HIDN, c16, 32 * kk, half);

#pragma unroll
    for (int t = 0; t < 8; ++t) {
      v8f acc = {};
#pragma unroll
      for (int kk = 0; kk < 4; ++kk) {
        v16h bf = loadMat16x32(W2t, HIDN, t * 16 + c16, 32 * kk, half);
        acc = __builtin_amdgcn_wmma_f32_16x16x32_f16(false, a2[kk].v, false, bf,
                                                     (short)0, acc, false, false);
      }
      const float bias = b2s[t * 16 + c16];
      float m = 0.f;                       // relu floor: max(relu(x)) = max(0, max(x))
#pragma unroll
      for (int r = 0; r < 8; ++r) m = fmaxf(m, acc[r] + bias);
      m = fmaxf(m, __shfl_xor(m, 16, 32)); // combine M halves (wave32)
      if (lane < 16) poolw[p * HIDN + t * 16 + c16] = (_Float16)m;
    }
  }

  // ---- layer 3: [16 points, 128] x [128,128] -> out
  FragU a3[4];
#pragma unroll
  for (int kk = 0; kk < 4; ++kk)
    a3[kk].v = loadMat16x32(poolw, HIDN, c16, 32 * kk, half);

#pragma unroll
  for (int t = 0; t < 8; ++t) {
    v8f acc = {};
#pragma unroll
    for (int kk = 0; kk < 4; ++kk) {
      v16h bf = loadMat16x32(W3t, HIDN, t * 16 + c16, 32 * kk, half);
      acc = __builtin_amdgcn_wmma_f32_16x16x32_f16(false, a3[kk].v, false, bf,
                                                   (short)0, acc, false, false);
    }
    const float bias = b3s[t * 16 + c16];
#pragma unroll
    for (int r = 0; r < 8; ++r) {
      out[((size_t)base + half * 8 + r) * HIDN + t * 16 + c16] = acc[r] + bias;
    }
  }
}

extern "C" void kernel_launch(void* const* d_in, const int* in_sizes, int n_in,
                              void* d_out, int out_size, void* d_ws, size_t ws_size,
                              hipStream_t stream) {
  (void)in_sizes; (void)n_in; (void)out_size; (void)d_ws; (void)ws_size;
  const float* f   = (const float*)d_in[0];
  const int*   knn = (const int*)d_in[1];
  const float* W1  = (const float*)d_in[2];
  const float* b1  = (const float*)d_in[3];
  const float* W2  = (const float*)d_in[4];
  const float* b2  = (const float*)d_in[5];
  const float* W3  = (const float*)d_in[6];
  const float* b3  = (const float*)d_in[7];
  float* out = (float*)d_out;

  hipFuncSetAttribute(reinterpret_cast<const void*>(knnconv_wmma_kernel),
                      hipFuncAttributeMaxDynamicSharedMemorySize, SMEM_BYTES);

  // 32768 points / (16 points per wave * 4 waves per block) = 512 blocks
  knnconv_wmma_kernel<<<dim3(512), dim3(128), SMEM_BYTES, stream>>>(
      f, knn, W1, b1, W2, b2, W3, b3, out);
}
